// VisionTransformer_36661840839774
// MI455X (gfx1250) — matrix operand order repaired and checked by reading, not compiled, not run on hardware
//
#include <hip/hip_runtime.h>

// ---------------------------------------------------------------------------
// CDNA5 / gfx1250 ViT forward.  All GEMMs via v_wmma_f32_16x16x32_f16.
// wave32; 64x128 block tile, 4 waves, each wave 32x64 (2x4 WMMA tiles ->
// 8 wmma per K-step).  LDS ping-pong staging with b128 loads/stores,
// register double buffering, one barrier per K-step.  All B operands
// pre-transposed to [N,K].
// ---------------------------------------------------------------------------

typedef _Float16 v16h __attribute__((ext_vector_type(16)));
typedef _Float16 half8 __attribute__((ext_vector_type(8)));
typedef float    v8f  __attribute__((ext_vector_type(8)));

#define BB 64       // batch
#define SS 197      // seq len
#define SP 224      // padded seq len (multiple of 32)
#define DD 1024     // token dim
#define EE 512      // per-head dim
#define HH 2        // heads
#define NPATCH 196
#define KPATCH 768  // 16*16*3
#define TT (BB * SS)     // 12608 tokens
#define TP (BB * NPATCH) // 12544 patch tokens
#define D2 2048
#define NCLS 1000

// ------------------------------- GEMM --------------------------------------

struct GemmP {
    const _Float16* A;     // [M,K] row major (per z)
    const _Float16* B;     // [N,K] row major (per z)  (always "transB")
    const float*    bias;  // [N] or null
    float*          Cf;    // optional f32 out
    _Float16*       Ch;    // optional f16 out
    int M, N, K, lda, ldb, ldc;
    int zdiv;              // z -> (zo = z/zdiv, zi = z%zdiv)
    long long sAo, sAi, sBo, sBi, sCo, sCi, sBiasI;
    int transC, relu;      // transC: Ch[n*ldc + m]
};

__device__ __forceinline__ void load_tiles(const _Float16* __restrict__ A,
                                           const _Float16* __restrict__ Bm,
                                           int m0, int n0, int M, int N,
                                           int lda, int ldb, int k0, int tid,
                                           uint4* ra, uint4* rb) {
#pragma unroll
    for (int it = 0; it < 2; ++it) {           // A: 64 rows x 4 segs
        const int i = it * 128 + tid;          // 0..255
        const int row = i >> 2, seg = i & 3;
        uint4 va = {0u, 0u, 0u, 0u};
        if (m0 + row < M)
            va = *(const uint4*)(A + (size_t)(m0 + row) * lda + k0 + seg * 8);
        ra[it] = va;
    }
#pragma unroll
    for (int it = 0; it < 4; ++it) {           // B: 128 rows x 4 segs
        const int i = it * 128 + tid;          // 0..511
        const int row = i >> 2, seg = i & 3;
        uint4 vb = {0u, 0u, 0u, 0u};
        if (n0 + row < N)
            vb = *(const uint4*)(Bm + (size_t)(n0 + row) * ldb + k0 + seg * 8);
        rb[it] = vb;
    }
}

__device__ __forceinline__ void store_tiles(_Float16 (*As)[40], _Float16 (*Bs)[40],
                                            int tid, const uint4* ra, const uint4* rb) {
#pragma unroll
    for (int it = 0; it < 2; ++it) {
        const int i = it * 128 + tid;
        const int row = i >> 2, seg = i & 3;
        *(uint4*)&As[row][seg * 8] = ra[it];
    }
#pragma unroll
    for (int it = 0; it < 4; ++it) {
        const int i = it * 128 + tid;
        const int row = i >> 2, seg = i & 3;
        *(uint4*)&Bs[row][seg * 8] = rb[it];
    }
}

__global__ __launch_bounds__(128) void gemm_f16_wmma(GemmP p) {
    __shared__ _Float16 AsBuf[2][64][40];    // ping-pong, 16B-aligned rows (80B stride)
    __shared__ _Float16 BsBuf[2][128][40];

    const int tid   = threadIdx.x;
    const int lane  = tid & 31;
    const int wv    = tid >> 5;
    const int waveM = wv & 1;       // 2 waves along M (32 rows each)
    const int waveN = wv >> 1;      // 2 waves along N (64 cols each)
    const int lhi   = lane >> 4;
    const int lm    = lane & 15;

    const int z  = blockIdx.z;
    const int zo = z / p.zdiv;
    const int zi = z % p.zdiv;
    const _Float16* A   = p.A + zo * p.sAo + zi * p.sAi;
    const _Float16* Bm  = p.B + zo * p.sBo + zi * p.sBi;
    const float*    bia = p.bias ? (p.bias + zi * p.sBiasI) : nullptr;
    const long long cOff = zo * p.sCo + zi * p.sCi;

    const int m0 = blockIdx.y * 64;
    const int n0 = blockIdx.x * 128;

    v8f acc[2][4] = {};
    uint4 ra[2], rb[4];

    load_tiles(A, Bm, m0, n0, p.M, p.N, p.lda, p.ldb, 0, tid, ra, rb);
    store_tiles(AsBuf[0], BsBuf[0], tid, ra, rb);

    const int nk = p.K >> 5;
    for (int kk = 0; kk < nk; ++kk) {
        const int  cur  = kk & 1;
        const bool more = (kk + 1) < nk;
        if (more)
            load_tiles(A, Bm, m0, n0, p.M, p.N, p.lda, p.ldb, (kk + 1) * 32, tid, ra, rb);
        if (kk + 2 < nk && tid < 64) {  // global_prefetch_b8 two tiles ahead
            const int m = m0 + tid;
            if (m < p.M) __builtin_prefetch(A + (size_t)m * p.lda + (kk + 2) * 32, 0, 0);
        }
        __syncthreads();

        _Float16 (*As)[40] = AsBuf[cur];
        _Float16 (*Bs)[40] = BsBuf[cur];
        v16h af[2], bf[4];
#pragma unroll
        for (int i = 0; i < 2; ++i) {
            const int row = waveM * 32 + i * 16 + lm;
            half8 lo = *(const half8*)&As[row][lhi * 8];        // k = lhi*8 + 0..7
            half8 hi = *(const half8*)&As[row][16 + lhi * 8];   // k = 16 + lhi*8 + 0..7
            af[i] = __builtin_shufflevector(lo, hi, 0,1,2,3,4,5,6,7,8,9,10,11,12,13,14,15);
        }
#pragma unroll
        for (int j = 0; j < 4; ++j) {
            const int col = waveN * 64 + j * 16 + lm;
            half8 lo = *(const half8*)&Bs[col][lhi * 8];        // k = lhi*8 + 0..7
            half8 hi = *(const half8*)&Bs[col][16 + lhi * 8];   // k = 16 + lhi*8 + 0..7
            bf[j] = __builtin_shufflevector(lo, hi, 0,1,2,3,4,5,6,7,8,9,10,11,12,13,14,15);
        }
#pragma unroll
        for (int i = 0; i < 2; ++i)
#pragma unroll
            for (int j = 0; j < 4; ++j)
                acc[i][j] = __builtin_amdgcn_wmma_f32_16x16x32_f16(
                    false, af[i], false, bf[j], (short)0, acc[i][j], false, false);

        if (more)
            store_tiles(AsBuf[cur ^ 1], BsBuf[cur ^ 1], tid, ra, rb);
    }

    // ---- epilogue: bias + relu, f32/f16 (optionally transposed f16) stores
    float*    Cf = p.Cf ? p.Cf + cOff : nullptr;
    _Float16* Ch = p.Ch ? p.Ch + cOff : nullptr;
#pragma unroll
    for (int i = 0; i < 2; ++i)
#pragma unroll
        for (int j = 0; j < 4; ++j)
#pragma unroll
            for (int v = 0; v < 8; ++v) {
                const int m = m0 + waveM * 32 + i * 16 + lhi * 8 + v;
                const int n = n0 + waveN * 64 + j * 16 + lm;
                if (m < p.M && n < p.N) {
                    float val = acc[i][j][v];
                    if (bia) val += bia[n];
                    if (p.relu && val < 0.f) val = 0.f;
                    if (Cf) Cf[(size_t)m * p.ldc + n] = val;
                    if (Ch) {
                        if (p.transC) Ch[(size_t)n * p.ldc + m] = (_Float16)val;
                        else          Ch[(size_t)m * p.ldc + n] = (_Float16)val;
                    }
                }
            }
}

// --------------------------- helper kernels --------------------------------

__global__ void cvt_h(const float* __restrict__ in, _Float16* __restrict__ out, long long n) {
    long long i = (long long)blockIdx.x * 256 + threadIdx.x;
    if (i < n) out[i] = (_Float16)in[i];
}

// in [Z,K,N] (f32) -> out [Z,N,K] (f16)
__global__ void cvt_t_h(const float* __restrict__ in, _Float16* __restrict__ out,
                        int Z, int K, int N) {
    long long idx = (long long)blockIdx.x * 256 + threadIdx.x;
    const long long total = (long long)Z * K * N;
    if (idx >= total) return;
    const int k = (int)(idx % K);
    const int n = (int)((idx / K) % N);
    const int h = (int)(idx / ((long long)K * N));
    out[idx] = (_Float16)in[((size_t)h * K + k) * N + n];
}

__global__ void patchify_h(const float* __restrict__ x, _Float16* __restrict__ out) {
    long long idx = (long long)blockIdx.x * 256 + threadIdx.x;
    const long long total = (long long)TP * KPATCH;
    if (idx >= total) return;
    const int k = (int)(idx % KPATCH);
    const long long t = idx / KPATCH;
    const int n = (int)(t % NPATCH);
    const int b = (int)(t / NPATCH);
    const int hh = n / 14, ww = n % 14;
    const int p1 = k / 48, rem = k % 48, p2 = rem / 3, c = rem % 3;
    const size_t src = (((size_t)b * 3 + c) * 224 + hh * 16 + p1) * 224 + ww * 16 + p2;
    out[idx] = (_Float16)x[src];
}

__global__ void build_seq(const float* __restrict__ h1, const float* __restrict__ pos,
                          const float* __restrict__ cls, float* __restrict__ seq) {
    long long idx = (long long)blockIdx.x * 256 + threadIdx.x;
    const long long total = (long long)TT * DD;
    if (idx >= total) return;
    const int d = (int)(idx % DD);
    const long long t = idx / DD;
    const int s = (int)(t % SS);
    const int b = (int)(t / SS);
    float v;
    if (s == 0) v = (d < EE) ? cls[d] : pos[d - EE];
    else        v = (d < EE) ? h1[((size_t)b * NPATCH + s - 1) * EE + d]
                             : pos[(size_t)s * EE + (d - EE)];
    seq[idx] = v;
}

__global__ void copy_skip(const float* __restrict__ seq, float* __restrict__ z) {
    long long idx = (long long)blockIdx.x * 256 + threadIdx.x;
    const long long total = (long long)TT * DD;
    if (idx >= total) return;
    z[(idx / DD) * D2 + (idx % DD)] = seq[idx];
}

// BatchNorm1d(S): normalize each seq position over (batch, feature); out f16.
__global__ __launch_bounds__(256) void bn_seq(const float* __restrict__ x,
                                              const float* __restrict__ g,
                                              const float* __restrict__ b,
                                              _Float16* __restrict__ out, int Dd) {
    const int s = blockIdx.x;
    __shared__ float r1[256], r2[256];
    const int total = BB * Dd;
    float sum = 0.f, sq = 0.f;
    for (int i = threadIdx.x; i < total; i += 256) {
        const int bb = i / Dd, d = i % Dd;
        const float v = x[((size_t)bb * SS + s) * Dd + d];
        sum += v; sq += v * v;
    }
    r1[threadIdx.x] = sum; r2[threadIdx.x] = sq;
    __syncthreads();
    for (int off = 128; off > 0; off >>= 1) {
        if (threadIdx.x < off) { r1[threadIdx.x] += r1[threadIdx.x + off];
                                 r2[threadIdx.x] += r2[threadIdx.x + off]; }
        __syncthreads();
    }
    const float mean = r1[0] / total;
    const float var  = r2[0] / total - mean * mean;
    const float inv  = rsqrtf(var + 1e-5f) * g[s];
    const float beta = b[s];
    for (int i = threadIdx.x; i < total; i += 256) {
        const int bb = i / Dd, d = i % Dd;
        const size_t idx = ((size_t)bb * SS + s) * Dd + d;
        out[idx] = (_Float16)((x[idx] - mean) * inv + beta);
    }
}

// softmax over valid 197 cols, 1/sqrt(1024) scale folded in; zero pad cols.
__global__ __launch_bounds__(256) void softmax_row(const float* __restrict__ sc,
                                                   _Float16* __restrict__ out) {
    const int row = blockIdx.x;          // 0..196
    const int bh  = blockIdx.y;          // 0..127
    const float* r = sc  + ((size_t)bh * SP + row) * SP;
    _Float16*    o = out + ((size_t)bh * SP + row) * SP;
    __shared__ float red[256];
    const int t = threadIdx.x;
    const float v = (t < SS) ? r[t] * (1.f / 32.f) : -1e30f;
    red[t] = v; __syncthreads();
    for (int off = 128; off > 0; off >>= 1) {
        if (t < off) red[t] = fmaxf(red[t], red[t + off]);
        __syncthreads();
    }
    const float mx = red[0]; __syncthreads();
    const float e = (t < SS) ? __expf(v - mx) : 0.f;
    red[t] = e; __syncthreads();
    for (int off = 128; off > 0; off >>= 1) {
        if (t < off) red[t] += red[t + off];
        __syncthreads();
    }
    const float inv = 1.f / red[0];
    if (t < SP) o[t] = (_Float16)(e * inv);   // pad cols get 0
}

// zero columns [start, rowlen) of each row (for Vt pad columns)
__global__ void zero_tail(_Float16* p, long long nrows, int rowlen, int start) {
    long long idx = (long long)blockIdx.x * 256 + threadIdx.x;
    const int tail = rowlen - start;
    const long long total = nrows * tail;
    if (idx >= total) return;
    const long long r = idx / tail;
    const int c = start + (int)(idx % tail);
    p[r * rowlen + c] = (_Float16)0;
}

__global__ void gather_cls(const _Float16* __restrict__ in, _Float16* __restrict__ out) {
    const int i = blockIdx.x * 256 + threadIdx.x;    // 64*256
    if (i >= BB * 256) return;
    const int b = i >> 8, e = i & 255;
    out[i] = in[((size_t)b * SS) * 256 + e];
}

// ------------------------------ host side ----------------------------------

static void launch_gemm(hipStream_t st, const _Float16* A, const _Float16* B,
                        const float* bias, float* Cf, _Float16* Ch,
                        int M, int N, int K, int lda, int ldb, int ldc,
                        int Z, int zdiv,
                        long long sAo, long long sAi, long long sBo, long long sBi,
                        long long sCo, long long sCi, long long sBiasI,
                        int transC, int relu) {
    GemmP p{A, B, bias, Cf, Ch, M, N, K, lda, ldb, ldc, zdiv,
            sAo, sAi, sBo, sBi, sCo, sCi, sBiasI, transC, relu};
    dim3 g((N + 127) / 128, (M + 63) / 64, Z);
    gemm_f16_wmma<<<g, 128, 0, st>>>(p);
}

extern "C" void kernel_launch(void* const* d_in, const int* in_sizes, int n_in,
                              void* d_out, int out_size, void* d_ws, size_t ws_size,
                              hipStream_t stream) {
    (void)in_sizes; (void)n_in; (void)out_size; (void)ws_size;
    const float* x      = (const float*)d_in[0];
    const float* W_pre  = (const float*)d_in[1];
    const float* b_pre  = (const float*)d_in[2];
    const float* pos    = (const float*)d_in[3];
    const float* cls    = (const float*)d_in[4];
    const float* Qw     = (const float*)d_in[5];
    const float* Qb     = (const float*)d_in[6];
    const float* Kw     = (const float*)d_in[7];
    const float* Kb     = (const float*)d_in[8];
    const float* Vw     = (const float*)d_in[9];
    const float* Vb     = (const float*)d_in[10];
    const float* bn1_g  = (const float*)d_in[11];
    const float* bn1_b  = (const float*)d_in[12];
    const float* bn2_g  = (const float*)d_in[13];
    const float* bn2_b  = (const float*)d_in[14];
    const float* W1     = (const float*)d_in[15];
    const float* b1     = (const float*)d_in[16];
    const float* W2     = (const float*)d_in[17];
    const float* b2     = (const float*)d_in[18];
    const float* W_post = (const float*)d_in[19];
    const float* b_post = (const float*)d_in[20];
    float* out = (float*)d_out;

    // ---- carve workspace (256B aligned blocks)
    char* w = (char*)d_ws;
    size_t off = 0;
    auto alloc = [&](size_t bytes) -> void* {
        off = (off + 255) & ~(size_t)255;
        void* r = w + off;
        off += bytes;
        return r;
    };
    _Float16* WpreT   = (_Float16*)alloc((size_t)EE * KPATCH * 2);     // [512,768]
    _Float16* QwT     = (_Float16*)alloc((size_t)HH * EE * DD * 2);    // [2,512,1024]
    _Float16* KwT     = (_Float16*)alloc((size_t)HH * EE * DD * 2);
    _Float16* VwT     = (_Float16*)alloc((size_t)HH * EE * DD * 2);
    _Float16* W1T     = (_Float16*)alloc((size_t)EE * D2 * 2);         // [512,2048]
    _Float16* W2T     = (_Float16*)alloc((size_t)256 * EE * 2);        // [256,512]
    _Float16* WpostT  = (_Float16*)alloc((size_t)NCLS * 256 * 2);      // [1000,256]
    _Float16* P_h     = (_Float16*)alloc((size_t)TP * KPATCH * 2);
    float*    h1      = (float*)   alloc((size_t)TP * EE * 4);
    float*    seq     = (float*)   alloc((size_t)TT * DD * 4);
    _Float16* y_h     = (_Float16*)alloc((size_t)TT * DD * 2);
    _Float16* Qbuf    = (_Float16*)alloc((size_t)BB * HH * SP * EE * 2);
    _Float16* Kbuf    = (_Float16*)alloc((size_t)BB * HH * SP * EE * 2);
    _Float16* VbufT   = (_Float16*)alloc((size_t)BB * HH * EE * SP * 2); // [bh, E, Sp]
    float*    sc_f    = (float*)   alloc((size_t)BB * HH * SP * SP * 4);
    _Float16* sc_h    = (_Float16*)alloc((size_t)BB * HH * SP * SP * 2);
    float*    z_f     = (float*)   alloc((size_t)TT * D2 * 4);
    _Float16* z_h     = (_Float16*)alloc((size_t)TT * D2 * 2);
    _Float16* ff1_h   = (_Float16*)alloc((size_t)TT * EE * 2);
    _Float16* ff2_h   = (_Float16*)alloc((size_t)TT * 256 * 2);
    _Float16* cls_h   = (_Float16*)alloc((size_t)BB * 256 * 2);

    auto blks = [](long long n) { return (unsigned)((n + 255) / 256); };

    // ---- weight conversions f32 -> f16, transposed to [N,K]
    cvt_t_h<<<blks((long long)KPATCH * EE), 256, 0, stream>>>(W_pre, WpreT, 1, KPATCH, EE);
    cvt_t_h<<<blks((long long)HH * DD * EE), 256, 0, stream>>>(Qw, QwT, HH, DD, EE);
    cvt_t_h<<<blks((long long)HH * DD * EE), 256, 0, stream>>>(Kw, KwT, HH, DD, EE);
    cvt_t_h<<<blks((long long)HH * DD * EE), 256, 0, stream>>>(Vw, VwT, HH, DD, EE);
    cvt_t_h<<<blks((long long)D2 * EE), 256, 0, stream>>>(W1, W1T, 1, D2, EE);
    cvt_t_h<<<blks((long long)EE * 256), 256, 0, stream>>>(W2, W2T, 1, EE, 256);
    cvt_t_h<<<blks((long long)256 * NCLS), 256, 0, stream>>>(W_post, WpostT, 1, 256, NCLS);

    // ---- patchify + patch projection (relu)
    patchify_h<<<blks((long long)TP * KPATCH), 256, 0, stream>>>(x, P_h);
    launch_gemm(stream, P_h, WpreT, b_pre, h1, nullptr,
                TP, EE, KPATCH, KPATCH, KPATCH, EE,
                1, 1, 0,0, 0,0, 0,0, 0, 0, 1);

    // ---- build seq [B,197,1024]; copy into skip half of z
    build_seq<<<blks((long long)TT * DD), 256, 0, stream>>>(h1, pos, cls, seq);
    copy_skip<<<blks((long long)TT * DD), 256, 0, stream>>>(seq, z_f);

    // ---- BN1 -> y (f16)
    bn_seq<<<SS, 256, 0, stream>>>(seq, bn1_g, bn1_b, y_h, DD);

    // ---- QKV projections: batched over (b,h); A per b, B/bias per h
    const long long sQC_o = (long long)HH * SP * EE, sQC_i = (long long)SP * EE;
    launch_gemm(stream, y_h, QwT, Qb, nullptr, Qbuf,
                SS, EE, DD, DD, DD, EE, BB * HH, HH,
                (long long)SS * DD, 0, 0, (long long)EE * DD,
                sQC_o, sQC_i, EE, 0, 0);
    launch_gemm(stream, y_h, KwT, Kb, nullptr, Kbuf,
                SS, EE, DD, DD, DD, EE, BB * HH, HH,
                (long long)SS * DD, 0, 0, (long long)EE * DD,
                sQC_o, sQC_i, EE, 0, 0);
    // V stored transposed: VbufT[bh][e][s], ldc = SP
    launch_gemm(stream, y_h, VwT, Vb, nullptr, VbufT,
                SS, EE, DD, DD, DD, SP, BB * HH, HH,
                (long long)SS * DD, 0, 0, (long long)EE * DD,
                (long long)HH * EE * SP, (long long)EE * SP, EE, 1, 0);
    // zero Vt pad columns (seq positions 197..223)
    zero_tail<<<blks((long long)BB * HH * EE * (SP - SS)), 256, 0, stream>>>(
        VbufT, (long long)BB * HH * EE, SP, SS);

    // ---- scores = Q @ K^T, per (b,h)   (B operand = K, already [N,K] form)
    launch_gemm(stream, Qbuf, Kbuf, nullptr, sc_f, nullptr,
                SS, SS, EE, EE, EE, SP, BB * HH, 1,
                (long long)SP * EE, 0, (long long)SP * EE, 0,
                (long long)SP * SP, 0, 0, 0, 0);

    // ---- softmax (scale 1/32 folded), zero pad cols, f16 out
    softmax_row<<<dim3(SS, BB * HH), 256, 0, stream>>>(sc_f, sc_h);

    // ---- attn = softmax @ V  (B operand = Vt [E, Sp]), into z[:,1024+h*512:]
    launch_gemm(stream, sc_h, VbufT, nullptr, z_f + DD, nullptr,
                SS, EE, SP, SP, SP, D2, BB * HH, HH,
                (long long)HH * SP * SP, (long long)SP * SP,
                (long long)HH * EE * SP, (long long)EE * SP,
                (long long)SS * D2, (long long)EE, 0, 0, 0);

    // ---- BN2 over z [B,197,2048] -> z_h (f16)
    bn_seq<<<SS, 256, 0, stream>>>(z_f, bn2_g, bn2_b, z_h, D2);

    // ---- MLP: W1 (relu), W2 (relu)
    launch_gemm(stream, z_h, W1T, b1, nullptr, ff1_h,
                TT, EE, D2, D2, D2, EE, 1, 1, 0,0, 0,0, 0,0, 0, 0, 1);
    launch_gemm(stream, ff1_h, W2T, b2, nullptr, ff2_h,
                TT, 256, EE, EE, EE, 256, 1, 1, 0,0, 0,0, 0,0, 0, 0, 1);

    // ---- classifier on CLS tokens
    gather_cls<<<blks((long long)BB * 256), 256, 0, stream>>>(ff2_h, cls_h);
    launch_gemm(stream, cls_h, WpostT, b_post, out, nullptr,
                BB, NCLS, 256, 256, 256, NCLS, 1, 1, 0,0, 0,0, 0,0, 0, 0, 0);
}
